// FixedRL_68968584839905
// MI455X (gfx1250) — compile-verified, matrix operand order checked by reading, not run on hardware
//
#include <hip/hip_runtime.h>

// ---------------------------------------------------------------------------
// Problem constants (fixed by the reference)
// ---------------------------------------------------------------------------
#define B_      64
#define U_      256
#define S_      64
#define DU_     16
#define DS_     16
#define H_      256
#define NU_     16384          // B*U
#define NS_     4096           // B*S
#define IN_DIM_ 5120           // U*DU + S*DS  (server block first: cols 0..1023)
#define NOUT_   512            // both heads' H fused side by side

// workspace layout (bytes, all 256-aligned)
#define OFF_W1T  0u                     // 512*5120 f16 = 5,242,880
#define OFF_W2T  5242880u               // 512*256  f16 =   262,144
#define OFF_H1   5505024u               // 16384*512 f16 = 16,777,216
#define OFF_H2   22282240u              // 16384*512 f16 = 16,777,216
#define OFF_MASK 39059456u              // 16384 u64     =    131,072
#define OFF_XU16 39190528u              // 16384*16 f16  =    524,288
#define OFF_XS16 39714816u              // 4096*16  f16  =    131,072
                                        // total ~ 39.8 MB

typedef __attribute__((ext_vector_type(16))) _Float16 v16h;
typedef __attribute__((ext_vector_type(8)))  _Float16 v8h;
typedef __attribute__((ext_vector_type(8)))  float    v8f;
typedef __attribute__((ext_vector_type(4)))  int      v4i;

#define GLOBAL_AS __attribute__((address_space(1)))
#define LDS_AS    __attribute__((address_space(3)))

// Load a 16x32 f16 WMMA fragment row-chunk for this lane.
// Per ISA 7.12.2: elements 0..7 -> K = half*8 + 0..7, elements 8..15 -> K = 16 + half*8 + 0..7.
// Caller passes p = rowBase + half*8 (+k0); chunks are 16B-aligned.
__device__ __forceinline__ v16h load_frag(const _Float16* p) {
  union { v16h v; v8h h[2]; } u;
  u.h[0] = *(const v8h*)(p);
  u.h[1] = *(const v8h*)(p + 16);
  return u.v;
}

__device__ __forceinline__ v8f wmma_f16(v16h a, v16h b, v8f c) {
  // (neg_a, A, neg_b, B, c_mod, C, reuse_a, reuse_b)
  return __builtin_amdgcn_wmma_f32_16x16x32_f16(false, a, false, b, (short)0, c,
                                                false, false);
}

// 16-byte async global->LDS copy (GLOBAL_LOAD_ASYNC_TO_LDS_B128, ASYNCcnt)
__device__ __forceinline__ void async_copy16(const _Float16* g, _Float16* l) {
  __builtin_amdgcn_global_load_async_to_lds_b128(
      (GLOBAL_AS v4i*)(void*)g, (LDS_AS v4i*)(void*)l, 0, 0);
}

// ---------------------------------------------------------------------------
// Input prep: convert x_user / x_server to f16 once so the per-tile scatter
// can be raw byte-moving async DMA.
// ---------------------------------------------------------------------------
__global__ void __launch_bounds__(256) k_prep_x16(const float* __restrict__ xu,
                                                  const float* __restrict__ xs,
                                                  _Float16* __restrict__ xu16,
                                                  _Float16* __restrict__ xs16) {
  int i = blockIdx.x * 256 + threadIdx.x;          // 1280 blocks -> 327680, exact
  if (i < NU_ * DU_) xu16[i] = (_Float16)xu[i];
  else               xs16[i - NU_ * DU_] = (_Float16)xs[i - NU_ * DU_];
}

// ---------------------------------------------------------------------------
// Weight prep: transpose + f32->f16, fusing o-head (n<256) and s-head (n>=256)
// ---------------------------------------------------------------------------
__global__ void __launch_bounds__(256) k_prep_w1t(const float* __restrict__ oW1,
                                                  const float* __restrict__ sW1,
                                                  _Float16* __restrict__ W1t) {
  int i = blockIdx.x * 256 + threadIdx.x;          // i < 512*5120, exact grid
  int n = i / IN_DIM_, k = i - n * IN_DIM_;
  float v = (n < H_) ? oW1[k * H_ + n] : sW1[k * H_ + (n - H_)];
  W1t[i] = (_Float16)v;
}

__global__ void __launch_bounds__(256) k_prep_w2t(const float* __restrict__ oW2,
                                                  const float* __restrict__ sW2,
                                                  _Float16* __restrict__ W2t) {
  int i = blockIdx.x * 256 + threadIdx.x;          // i < 512*256, exact grid
  int n = i >> 8, k = i & 255;
  float v = (n < H_) ? oW2[k * H_ + n] : sW2[k * H_ + (n - H_)];
  W2t[i] = (_Float16)v;
}

// ---------------------------------------------------------------------------
// Selection mask: bit s set iff user has a u2s edge to local server s
// ---------------------------------------------------------------------------
__global__ void __launch_bounds__(256) k_mask_zero(unsigned long long* __restrict__ mask) {
  mask[blockIdx.x * 256 + threadIdx.x] = 0ull;     // 64 blocks, exact
}

__global__ void __launch_bounds__(256) k_mask_or(const int* __restrict__ src,
                                                 const int* __restrict__ dst,
                                                 unsigned long long* __restrict__ mask) {
  int e = blockIdx.x * 256 + threadIdx.x;          // 512 blocks = NU*DEG_S, exact
  atomicOr(&mask[src[e]], 1ull << (dst[e] & 63));
}

// ---------------------------------------------------------------------------
// GEMM1: per block, build the dense 16-user observation tile in LDS
// (160 KB of the 320 KB WGP budget; feats never touches HBM). The scatter
// is done with GLOBAL_LOAD_ASYNC_TO_LDS_B128 (per-lane global addr ->
// per-lane LDS addr, ASYNCcnt), then
// h1[16x512] = relu(feats_tile @ W1t^T + b1) via v_wmma_f32_16x16x32_f16.
// 8 waves; wave w covers output columns [w*64, w*64+64): 4 N-tiles reusing
// one A fragment per k-step.
// ---------------------------------------------------------------------------
__global__ void __launch_bounds__(256) k_gemm1(
    const _Float16* __restrict__ xu16, const _Float16* __restrict__ xs16,
    const int* __restrict__ dst_u2s, const int* __restrict__ dst_u2u,
    const _Float16* __restrict__ W1t, const float* __restrict__ ob1,
    const float* __restrict__ sb1, _Float16* __restrict__ h1) {
  extern __shared__ _Float16 smA[];                // [16][IN_DIM_]
  const int tid = threadIdx.x;
  const int u0 = blockIdx.x * 16;

  // zero the tile: 16*5120 halves = 40960 dwords
  uint32_t* smz = (uint32_t*)smA;
  #pragma unroll
  for (int i = 0; i < 160; ++i) smz[tid + i * 256] = 0u;
  __syncthreads();

  // scatter u2s edges: 128 edges x 32B = 256 async b128 chunks (one/thread)
  {
    int e = tid >> 1, hb = (tid & 1) * 8;          // e = local edge id (ul*8+j)
    int ul = e >> 3;
    int d = dst_u2s[u0 * 8 + e];                   // global server id
    async_copy16(xs16 + d * 16 + hb, smA + ul * IN_DIM_ + (d & 63) * 16 + hb);
  }
  // scatter u2u edges: 256 edges x 32B = 512 async b128 chunks (two/thread)
  #pragma unroll
  for (int c0 = 0; c0 < 512; c0 += 256) {
    int c = c0 + tid;
    int e = c >> 1, hb = (c & 1) * 8;              // e = local edge id (ul*16+j)
    int ul = e >> 4;
    int d = dst_u2u[u0 * 16 + e];                  // global user id
    async_copy16(xu16 + d * 16 + hb,
                 smA + ul * IN_DIM_ + 1024 + (d & 255) * 16 + hb);
  }
  asm volatile("s_wait_asynccnt 0x0" ::: "memory");
  __syncthreads();

  const int wave = tid >> 5, lane = tid & 31;
  const int half = lane >> 4, nl = lane & 15;
  const int nbase = wave * 64;

  const _Float16* pa  = smA + nl * IN_DIM_ + half * 8;            // LDS (ds_load_b128)
  const _Float16* pb0 = W1t + (nbase + nl) * IN_DIM_ + half * 8;  // global
  const _Float16* pb1 = pb0 + 16 * IN_DIM_;
  const _Float16* pb2 = pb0 + 32 * IN_DIM_;
  const _Float16* pb3 = pb0 + 48 * IN_DIM_;

  v8f acc0 = {}, acc1 = {}, acc2 = {}, acc3 = {};
  #pragma unroll 2
  for (int k0 = 0; k0 < IN_DIM_; k0 += 32) {
    __builtin_prefetch(pb0 + k0 + 512, 0, 3);      // global_prefetch_b8 (WGP scope)
    v16h a = load_frag(pa + k0);
    acc0 = wmma_f16(a, load_frag(pb0 + k0), acc0);
    acc1 = wmma_f16(a, load_frag(pb1 + k0), acc1);
    acc2 = wmma_f16(a, load_frag(pb2 + k0), acc2);
    acc3 = wmma_f16(a, load_frag(pb3 + k0), acc3);
  }

  v8f accs[4] = {acc0, acc1, acc2, acc3};
  #pragma unroll
  for (int t = 0; t < 4; ++t) {
    int n = nbase + t * 16 + nl;
    float bias = (n < H_) ? ob1[n] : sb1[n - H_];
    #pragma unroll
    for (int v = 0; v < 8; ++v) {
      int m = v + 8 * half;                        // C/D layout: row = v + 8*half
      float x = accs[t][v] + bias;
      x = x > 0.f ? x : 0.f;                       // relu
      h1[(u0 + m) * NOUT_ + n] = (_Float16)x;
    }
  }
}

// ---------------------------------------------------------------------------
// GEMM2 (block-diagonal): h2[:,0:256]=sigmoid(h1[:,0:256]@oW2+b),
//                         h2[:,256:512]=sigmoid(h1[:,256:512]@sW2+b)
// One 16x16 tile per wave, K=256 (8 wmma steps, fully unrolled).
// ---------------------------------------------------------------------------
__global__ void __launch_bounds__(256) k_gemm2(
    const _Float16* __restrict__ h1, const _Float16* __restrict__ W2t,
    const float* __restrict__ ob2, const float* __restrict__ sb2,
    _Float16* __restrict__ h2) {
  const int wave = threadIdx.x >> 5, lane = threadIdx.x & 31;
  const int tile = blockIdx.x * 8 + wave;          // 32768 tiles over 4096 blocks
  const int mtile = tile >> 5, ntile = tile & 31;
  const int half = lane >> 4, nl = lane & 15;
  const int n = ntile * 16 + nl;
  const int koff = (ntile < 16) ? 0 : H_;          // which half of h1 feeds this head

  const _Float16* pa = h1 + (mtile * 16 + nl) * NOUT_ + koff + half * 8;
  const _Float16* pb = W2t + n * H_ + half * 8;

  v8f acc = {};
  #pragma unroll
  for (int k0 = 0; k0 < H_; k0 += 32)
    acc = wmma_f16(load_frag(pa + k0), load_frag(pb + k0), acc);

  float bias = (ntile < 16) ? ob2[n] : sb2[n - H_];
  #pragma unroll
  for (int v = 0; v < 8; ++v) {
    int m = v + 8 * half;
    float x = acc[v] + bias;
    x = 1.0f / (1.0f + __expf(-x));                // sigmoid
    h2[(mtile * 16 + m) * NOUT_ + n] = (_Float16)x;
  }
}

// ---------------------------------------------------------------------------
// Head 1: offloading logits (H->2) + softmax-2. One thread per user.
// ---------------------------------------------------------------------------
__global__ void __launch_bounds__(256) k_head_offload(
    const _Float16* __restrict__ h2, const float* __restrict__ oW3,
    const float* __restrict__ ob3, float* __restrict__ out) {
  int u = blockIdx.x * 256 + threadIdx.x;          // 64 blocks, exact
  const _Float16* hp = h2 + u * NOUT_;             // o-head half of h2
  float l0 = ob3[0], l1 = ob3[1];
  for (int k = 0; k < H_; ++k) {
    float a = (float)hp[k];
    l0 += a * oW3[k * 2 + 0];
    l1 += a * oW3[k * 2 + 1];
  }
  float m = fmaxf(l0, l1);
  float e0 = __expf(l0 - m), e1 = __expf(l1 - m);
  float s = e0 + e1;
  out[u * 2 + 0] = e0 / s;
  out[u * 2 + 1] = e1 / s;
}

// ---------------------------------------------------------------------------
// Head 2: selection logits (H->64) + edge mask + softmax-64.
// One wave32 per user; lane owns columns n and n+32; shfl reductions.
// ---------------------------------------------------------------------------
__global__ void __launch_bounds__(256) k_head_select(
    const _Float16* __restrict__ h2, const float* __restrict__ sW3,
    const float* __restrict__ sb3, const unsigned long long* __restrict__ mask,
    float* __restrict__ out) {
  const int wave = threadIdx.x >> 5, lane = threadIdx.x & 31;
  const int u = blockIdx.x * 8 + wave;             // 2048 blocks, exact
  const _Float16* hp = h2 + u * NOUT_ + H_;        // s-head half of h2
  float a0 = sb3[lane], a1 = sb3[lane + 32];
  for (int k = 0; k < H_; ++k) {
    float a = (float)hp[k];
    a0 += a * sW3[k * S_ + lane];
    a1 += a * sW3[k * S_ + lane + 32];
  }
  unsigned long long mk = mask[u];
  const float NEG = -__builtin_inff();
  if (!((mk >> lane) & 1ull)) a0 = NEG;
  if (!((mk >> (lane + 32)) & 1ull)) a1 = NEG;

  float m = fmaxf(a0, a1);
  #pragma unroll
  for (int off = 16; off >= 1; off >>= 1) m = fmaxf(m, __shfl_xor(m, off, 32));
  float e0 = __expf(a0 - m), e1 = __expf(a1 - m);
  float s = e0 + e1;
  #pragma unroll
  for (int off = 16; off >= 1; off >>= 1) s += __shfl_xor(s, off, 32);

  out[NU_ * 2 + u * S_ + lane]      = e0 / s;
  out[NU_ * 2 + u * S_ + lane + 32] = e1 / s;
}

// ---------------------------------------------------------------------------
extern "C" void kernel_launch(void* const* d_in, const int* in_sizes, int n_in,
                              void* d_out, int out_size, void* d_ws, size_t ws_size,
                              hipStream_t stream) {
  (void)in_sizes; (void)n_in; (void)out_size; (void)ws_size;

  const float* x_user   = (const float*)d_in[0];
  const float* x_server = (const float*)d_in[1];
  const int*   src_u2s  = (const int*)d_in[2];
  const int*   dst_u2s  = (const int*)d_in[3];
  const int*   dst_u2u  = (const int*)d_in[5];   // src_u2u (d_in[4]) implied by layout
  const float* oW1 = (const float*)d_in[6];
  const float* ob1 = (const float*)d_in[7];
  const float* oW2 = (const float*)d_in[8];
  const float* ob2 = (const float*)d_in[9];
  const float* oW3 = (const float*)d_in[10];
  const float* ob3 = (const float*)d_in[11];
  const float* sW1 = (const float*)d_in[12];
  const float* sb1 = (const float*)d_in[13];
  const float* sW2 = (const float*)d_in[14];
  const float* sb2 = (const float*)d_in[15];
  const float* sW3 = (const float*)d_in[16];
  const float* sb3 = (const float*)d_in[17];

  float* out = (float*)d_out;
  char*  ws  = (char*)d_ws;
  _Float16* W1t  = (_Float16*)(ws + OFF_W1T);
  _Float16* W2t  = (_Float16*)(ws + OFF_W2T);
  _Float16* h1   = (_Float16*)(ws + OFF_H1);
  _Float16* h2   = (_Float16*)(ws + OFF_H2);
  _Float16* xu16 = (_Float16*)(ws + OFF_XU16);
  _Float16* xs16 = (_Float16*)(ws + OFF_XS16);
  unsigned long long* mask = (unsigned long long*)(ws + OFF_MASK);

  k_prep_x16<<<1280, 256, 0, stream>>>(x_user, x_server, xu16, xs16);
  k_prep_w1t<<<10240, 256, 0, stream>>>(oW1, sW1, W1t);
  k_prep_w2t<<<512, 256, 0, stream>>>(oW2, sW2, W2t);
  k_mask_zero<<<64, 256, 0, stream>>>(mask);
  k_mask_or<<<512, 256, 0, stream>>>(src_u2s, dst_u2s, mask);
  k_gemm1<<<1024, 256, 16 * IN_DIM_ * sizeof(_Float16), stream>>>(
      xu16, xs16, dst_u2s, dst_u2u, W1t, ob1, sb1, h1);
  k_gemm2<<<4096, 256, 0, stream>>>(h1, W2t, ob2, sb2, h2);
  k_head_offload<<<64, 256, 0, stream>>>(h2, oW3, ob3, out);
  k_head_select<<<2048, 256, 0, stream>>>(h2, sW3, sb3, mask, out);
}
